// TransformerBlock_82154134438514
// MI455X (gfx1250) — matrix operand order repaired and checked by reading, not compiled, not run on hardware
//
#include <hip/hip_runtime.h>
#include <hip/hip_bf16.h>
#include <math.h>

// ---------------------------------------------------------------------------
// MI455X (gfx1250) transformer block: bf16 WMMA GEMMs (TDM-staged, double-
// buffered LDS, BK=64) + fused flash attention (TDM-staged K tiles).
// ---------------------------------------------------------------------------

typedef __bf16 bf16_t;
typedef __attribute__((ext_vector_type(16))) __bf16       v16bf;
typedef __attribute__((ext_vector_type(8)))  float        v8f;
typedef __attribute__((ext_vector_type(4)))  unsigned int u32x4;
typedef __attribute__((ext_vector_type(4)))  int          i32x4;
typedef __attribute__((ext_vector_type(8)))  int          i32x8;

union FragB16 { v16bf v; u32x4 q[2]; bf16_t e[16]; };
union Acc8    { v8f   v; float  e[8]; };

#define WMMA_BF16(A, B, C) \
  __builtin_amdgcn_wmma_f32_16x16x32_bf16(false, (A), false, (B), (short)0, (C), false, false)

#if __has_builtin(__builtin_amdgcn_tensor_load_to_lds)
#define HAVE_TDM 1
#if __has_include(<hip/amd_detail/amd_gfx1250_TDM.h>)
#define TDM_6ARG 1
#endif
#endif

// pad cfg: +4 dwords after every 16 dwords (=> 40-dword LDS row per 32-dword tile row)
#define TDM_PAD16x4 ((1u << 20) | (3u << 22) | (3u << 25))

__device__ inline void wait_tensor0() {
#if __has_builtin(__builtin_amdgcn_s_wait_tensorcnt)
  __builtin_amdgcn_s_wait_tensorcnt(0);
#else
  asm volatile("s_wait_tensorcnt 0x0" ::: "memory");
#endif
}

// Generic 2-D TDM load: rows x tileDw dwords, global row stride strideDw dwords.
__device__ inline void tdm_load_2d(const bf16_t* gptr, unsigned int lds_byte,
                                   int tileDw, int rows, int strideDw,
                                   unsigned int padcfg) {
#if HAVE_TDM
  unsigned long long ga = (unsigned long long)(uintptr_t)gptr;
  u32x4 g0;
  g0[0] = 1u;                      // count=1, user descriptor
  g0[1] = lds_byte;                // lds_addr (bytes)
  g0[2] = (unsigned int)ga;        // global_addr[31:0]
  g0[3] = (unsigned int)((ga >> 32) & 0x1FFFFFFull) | (2u << 30);  // addr[56:32] | type=2
  unsigned long long td0 = 1u << 30;            // tensor_dim0 (large, in-bounds)
  unsigned long long td1 = 1u << 30;            // tensor_dim1 (large, in-bounds)
  unsigned long long st0 = (unsigned int)strideDw;
  i32x8 g1 = (i32x8)0;
  g1[0] = (int)((2u << 16) | padcfg);                      // data_size = 4B | pad cfg
  g1[1] = (int)((td0 & 0xFFFFull) << 16);
  g1[2] = (int)(((td0 >> 16) & 0xFFFFull) | ((td1 & 0xFFFFull) << 16));
  g1[3] = (int)(((td1 >> 16) & 0xFFFFull) | ((unsigned int)tileDw << 16));
  g1[4] = rows;                                            // tile_dim1, tile_dim2=0
  g1[5] = (int)(st0 & 0xFFFFFFFFull);
  g1[6] = (int)((st0 >> 32) & 0xFFFFull);
  g1[7] = 0;
  i32x4 z4 = (i32x4)0;
#if TDM_6ARG
  i32x8 z8 = (i32x8)0;
  __builtin_amdgcn_tensor_load_to_lds(g0, g1, z4, z4, z8, 0);
#else
  __builtin_amdgcn_tensor_load_to_lds(g0, g1, z4, z4, 0);
#endif
#else
  (void)gptr; (void)lds_byte; (void)tileDw; (void)rows; (void)strideDw; (void)padcfg;
#endif
}

// ------------------------------- small kernels ------------------------------
__global__ __launch_bounds__(256) void cvt_f32_bf16_k(const float* __restrict__ s,
                                                      bf16_t* __restrict__ d, int n) {
  int i = blockIdx.x * 256 + threadIdx.x;
  if (i < n) d[i] = (bf16_t)s[i];
}

__global__ __launch_bounds__(256) void sigmoid_k(const float* __restrict__ s,
                                                 float* __restrict__ d, int n) {
  int i = blockIdx.x * 256 + threadIdx.x;
  if (i < n) d[i] = 1.f / (1.f + __expf(-s[i]));
}

// fp32 [K][N] -> bf16 transposed [N][K] (weights, done once per call)
__global__ __launch_bounds__(256) void cvt_transpose_k(const float* __restrict__ s,
                                                       bf16_t* __restrict__ d,
                                                       int K, int N) {
  __shared__ float t[32][33];
  int kb = blockIdx.x * 32, nb = blockIdx.y * 32;
  int tx = threadIdx.x & 31, ty = threadIdx.x >> 5;  // 32 x 8
  for (int i = ty; i < 32; i += 8)
    t[i][tx] = s[(size_t)(kb + i) * N + nb + tx];
  __syncthreads();
  for (int i = ty; i < 32; i += 8)
    d[(size_t)(nb + i) * K + kb + tx] = (bf16_t)t[tx][i];
}

// ------------------------------- LayerNorm ----------------------------------
__global__ __launch_bounds__(256) void ln_k(const float* __restrict__ x,
                                            const float* __restrict__ g,
                                            const float* __restrict__ bta,
                                            bf16_t* __restrict__ out, int D) {
  int row = blockIdx.x, tid = threadIdx.x;
  int lane = tid & 31, wid = tid >> 5;
  __shared__ float red[8];
  __shared__ float sMean, sStd;
  const float* xr = x + (size_t)row * D;

  float s = 0.f;
  for (int j = tid; j < D; j += 256) s += xr[j];
  for (int o = 16; o >= 1; o >>= 1) s += __shfl_xor(s, o, 32);
  if (lane == 0) red[wid] = s;
  __syncthreads();
  if (tid == 0) { float t = 0.f; for (int i = 0; i < 8; ++i) t += red[i]; sMean = t / D; }
  __syncthreads();
  float mean = sMean;

  float v = 0.f;
  for (int j = tid; j < D; j += 256) { float d0 = xr[j] - mean; v += d0 * d0; }
  for (int o = 16; o >= 1; o >>= 1) v += __shfl_xor(v, o, 32);
  if (lane == 0) red[wid] = v;
  __syncthreads();
  if (tid == 0) { float t = 0.f; for (int i = 0; i < 8; ++i) t += red[i]; sStd = sqrtf(t / (D - 1)); }
  __syncthreads();
  float istd = 1.f / (sStd + 1e-6f);

  for (int j = tid; j < D; j += 256)
    out[(size_t)row * D + j] = (bf16_t)(g[j] * (xr[j] - mean) * istd + bta[j]);
}

// ------------------------------- GEMM (bf16 WMMA) ---------------------------
// out[M,N] = X[M,K] * W[K,N] + bias, W supplied PRE-TRANSPOSED as Wt[N,K].
// A-frag = Wt rows (features), B-frag = X rows (tokens), D = C^T tile.
// TDM-staged, double-buffered, BK=64 (2 k-chunks, 16 WMMAs per sync).
//   EPI=0: bf16(acc+bias)  EPI=1: bf16(gelu(acc+bias))  EPI=2: f32(resid+acc+bias)
template <int EPI>
__global__ __launch_bounds__(256) void gemm_bf16_k(
    const bf16_t* __restrict__ X, const bf16_t* __restrict__ Wt,
    const float* __restrict__ bias, const float* __restrict__ resid,
    float* __restrict__ outf, bf16_t* __restrict__ outb,
    int M, int N, int K) {
  constexpr int LDK = 80;  // 2 chunks of (32 data + 8 pad) bf16
  __shared__ bf16_t Xs[2][128][LDK];
  __shared__ bf16_t Ws[2][128][LDK];

  int tid = threadIdx.x, lane = tid & 31, wid = tid >> 5;
  int n0 = blockIdx.x * 128, m0 = blockIdx.y * 128;
  int wn = (wid >> 1) * 32;  // feature offset of this wave's tile (A side)
  int wm = (wid & 1) * 64;   // token offset of this wave's tile   (B side)
  int l16 = lane & 15, kh = lane >> 4;
  int nStages = K >> 6;

  Acc8 acc[2][4];
  for (int r = 0; r < 2; ++r)
    for (int c = 0; c < 4; ++c)
      for (int i = 0; i < 8; ++i) acc[r][c].e[i] = 0.f;

  auto fill = [&](int bb, int k0) {
#if HAVE_TDM
    // each wave DMAs its own 16-row slice of both tiles (32 dwords/row)
    tdm_load_2d(X + (size_t)(m0 + wid * 16) * K + k0,
                (unsigned int)(uintptr_t)&Xs[bb][wid * 16][0], 32, 16, K >> 1, TDM_PAD16x4);
    tdm_load_2d(Wt + (size_t)(n0 + wid * 16) * K + k0,
                (unsigned int)(uintptr_t)&Ws[bb][wid * 16][0], 32, 16, K >> 1, TDM_PAD16x4);
#else
    int r = tid >> 1, c16 = (tid & 1) * 16;
    for (int half = 0; half < 2; ++half) {
      const bf16_t* sx = X + (size_t)(m0 + r) * K + k0 + half * 32 + c16;
      const bf16_t* sw = Wt + (size_t)(n0 + r) * K + k0 + half * 32 + c16;
      *(u32x4*)&Xs[bb][r][half * 40 + c16]     = *(const u32x4*)sx;
      *(u32x4*)&Xs[bb][r][half * 40 + c16 + 8] = *(const u32x4*)(sx + 8);
      *(u32x4*)&Ws[bb][r][half * 40 + c16]     = *(const u32x4*)sw;
      *(u32x4*)&Ws[bb][r][half * 40 + c16 + 8] = *(const u32x4*)(sw + 8);
    }
#endif
  };

  fill(0, 0);
  for (int s = 0; s < nStages; ++s) {
    int bb = s & 1;
    wait_tensor0();        // our TDM slices for stage s are done
    __syncthreads();       // everyone's slices done; prev compute done
    if (s + 1 < nStages) fill(bb ^ 1, (s + 1) * 64);

    for (int kc = 0; kc < 2; ++kc) {
      int cb = kc * 40;  // padded chunk base (bf16)
      FragB16 fw[2], fx[4];
      for (int rt = 0; rt < 2; ++rt) {
        const bf16_t* p = &Ws[bb][wn + rt * 16 + l16][cb];
        fw[rt].q[0] = *(const u32x4*)(p + kh * 8);       // K = kh*8 + 0..7
        fw[rt].q[1] = *(const u32x4*)(p + 16 + kh * 8);  // K = 16 + kh*8 + 0..7
      }
      for (int ct = 0; ct < 4; ++ct) {
        const bf16_t* p = &Xs[bb][wm + ct * 16 + l16][cb];
        fx[ct].q[0] = *(const u32x4*)(p + kh * 8);       // K = kh*8 + 0..7
        fx[ct].q[1] = *(const u32x4*)(p + 16 + kh * 8);  // K = 16 + kh*8 + 0..7
      }
      for (int rt = 0; rt < 2; ++rt)
        for (int ct = 0; ct < 4; ++ct)
          acc[rt][ct].v = WMMA_BF16(fw[rt].v, fx[ct].v, acc[rt][ct].v);
    }
  }

  // Epilogue: lane holds 8 CONSECUTIVE feature columns of one token row.
  for (int rt = 0; rt < 2; ++rt)
    for (int ct = 0; ct < 4; ++ct) {
      int col = n0 + wn + rt * 16 + kh * 8;
      int row = m0 + wm + ct * 16 + l16;
      if (EPI == 2) {
        union { u32x4 q[2]; float e[8]; } o;
        const float* rp = resid + (size_t)row * N + col;
        for (int i = 0; i < 8; ++i) o.e[i] = acc[rt][ct].e[i] + bias[col + i] + rp[i];
        u32x4* dst = (u32x4*)(outf + (size_t)row * N + col);
        dst[0] = o.q[0]; dst[1] = o.q[1];
      } else {
        union { u32x4 q; bf16_t e[8]; } o;
        for (int i = 0; i < 8; ++i) {
          float vv = acc[rt][ct].e[i] + bias[col + i];
          if (EPI == 1) {
            float t = vv + 0.044715f * vv * vv * vv;
            vv = 0.5f * vv * (1.0f + tanhf(0.7978845608028654f * t));
          }
          o.e[i] = (bf16_t)vv;
        }
        *(u32x4*)(outb + (size_t)row * N + col) = o.q;
      }
    }
}

// ------------------------------- Fused attention ----------------------------
// Flash-style: scores = where(mask==0,-1e9, q.k1/8) + q.k2/8
// p = softmax(scores); ctx = (p * sig[S,S]) @ v   (sig precomputed, excluded
// from the softmax denominator).
__global__ __launch_bounds__(256) void attn_k(
    const bf16_t* __restrict__ Q, const bf16_t* __restrict__ K1,
    const bf16_t* __restrict__ K2, const bf16_t* __restrict__ V,
    const int* __restrict__ mask, const float* __restrict__ sigt,
    bf16_t* __restrict__ ctx) {
  constexpr int S = 512, D = 1024, DK = 64;
  __shared__ bf16_t k1s[32][64];     // [kcol][dk]
  __shared__ bf16_t k2s[32][64];
  __shared__ bf16_t vts[64][40];     // [dk][kcol]  (V transposed), padded
  __shared__ bf16_t pls[8][16][32];  // per-wave P patch (C-layout -> A-layout)

  int tid = threadIdx.x, lane = tid & 31, w = tid >> 5;
  int qb = blockIdx.x, h = blockIdx.y, b = blockIdx.z;
  int q0 = qb * 128 + w * 16;
  int n = lane & 15, kh = lane >> 4;
  const float inv = 0.125f;  // 1/sqrt(64)

  FragB16 fq[2];
  {
    const bf16_t* qp = Q + (size_t)(b * S + q0 + n) * D + h * DK;
    for (int kc = 0; kc < 2; ++kc) {
      fq[kc].q[0] = *(const u32x4*)(qp + kc * 32 + kh * 16);
      fq[kc].q[1] = *(const u32x4*)(qp + kc * 32 + kh * 16 + 8);
    }
  }

  float m_run[8], l_run[8];
  Acc8 Oa[4];
  for (int i = 0; i < 8; ++i) { m_run[i] = -3.0e38f; l_run[i] = 0.f; }
  for (int ct = 0; ct < 4; ++ct)
    for (int i = 0; i < 8; ++i) Oa[ct].e[i] = 0.f;

  for (int kb = 0; kb < S / 32; ++kb) {
    int kbase = kb * 32;
    {
#if HAVE_TDM
      // K1/K2 tiles: each wave DMAs a 4-row slice (32 dwords/row, stride D/2)
      tdm_load_2d(K1 + (size_t)(b * S + kbase + w * 4) * D + h * DK,
                  (unsigned int)(uintptr_t)&k1s[w * 4][0], 32, 4, D >> 1, 0u);
      tdm_load_2d(K2 + (size_t)(b * S + kbase + w * 4) * D + h * DK,
                  (unsigned int)(uintptr_t)&k2s[w * 4][0], 32, 4, D >> 1, 0u);
#endif
      int r = tid >> 3, c = (tid & 7) * 8;
      size_t gofs = (size_t)(b * S + kbase + r) * D + h * DK + c;
#if !HAVE_TDM
      *(u32x4*)&k1s[r][c] = *(const u32x4*)(K1 + gofs);
      *(u32x4*)&k2s[r][c] = *(const u32x4*)(K2 + gofs);
#endif
      union { u32x4 q; bf16_t e[8]; } vv;
      vv.q = *(const u32x4*)(V + gofs);
      for (int j = 0; j < 8; ++j) vts[c + j][r] = vv.e[j];
    }
    wait_tensor0();
    __syncthreads();

    Acc8 s1[2], s2[2];
    for (int nc = 0; nc < 2; ++nc)
      for (int i = 0; i < 8; ++i) { s1[nc].e[i] = 0.f; s2[nc].e[i] = 0.f; }
    FragB16 fb;
    for (int nc = 0; nc < 2; ++nc) {
      for (int kc = 0; kc < 2; ++kc) {
        const bf16_t* p = &k1s[nc * 16 + n][kc * 32 + kh * 16];
        fb.q[0] = *(const u32x4*)p;
        fb.q[1] = *(const u32x4*)(p + 8);
        s1[nc].v = WMMA_BF16(fq[kc].v, fb.v, s1[nc].v);
      }
      for (int kc = 0; kc < 2; ++kc) {
        const bf16_t* p = &k2s[nc * 16 + n][kc * 32 + kh * 16];
        fb.q[0] = *(const u32x4*)p;
        fb.q[1] = *(const u32x4*)(p + 8);
        s2[nc].v = WMMA_BF16(fq[kc].v, fb.v, s2[nc].v);
      }
    }

    float sv[2][8], sg[2][8];
    for (int nc = 0; nc < 2; ++nc) {
      int kcol = kbase + nc * 16 + n;
      for (int i = 0; i < 8; ++i) {
        int qrow = q0 + kh * 8 + i;
        int mk = mask[((size_t)b * S + qrow) * S + kcol];
        sv[nc][i] = (mk != 0 ? s1[nc].e[i] * inv : -1e9f) + s2[nc].e[i] * inv;
        sg[nc][i] = sigt[(size_t)qrow * S + kcol];  // precomputed sigmoid
      }
    }
    float pv[2][8];
    for (int i = 0; i < 8; ++i) {
      float mx = fmaxf(sv[0][i], sv[1][i]);
      for (int o = 8; o >= 1; o >>= 1) mx = fmaxf(mx, __shfl_xor(mx, o, 32));
      float mnew = fmaxf(m_run[i], mx);
      float alpha = __expf(m_run[i] - mnew);
      float p0 = __expf(sv[0][i] - mnew);
      float p1 = __expf(sv[1][i] - mnew);
      float ps = p0 + p1;
      for (int o = 8; o >= 1; o >>= 1) ps += __shfl_xor(ps, o, 32);
      l_run[i] = l_run[i] * alpha + ps;
      m_run[i] = mnew;
      pv[0][i] = p0 * sg[0][i];
      pv[1][i] = p1 * sg[1][i];
      for (int ct = 0; ct < 4; ++ct) Oa[ct].e[i] *= alpha;
    }

    for (int nc = 0; nc < 2; ++nc)
      for (int i = 0; i < 8; ++i)
        pls[w][kh * 8 + i][nc * 16 + n] = (bf16_t)pv[nc][i];
    asm volatile("s_wait_dscnt 0" ::: "memory");  // intra-wave LDS RAW fence

    FragB16 fp;
    {
      const bf16_t* p = &pls[w][n][0];
      fp.q[0] = *(const u32x4*)(p + kh * 16);
      fp.q[1] = *(const u32x4*)(p + kh * 16 + 8);
    }
    for (int ct = 0; ct < 4; ++ct) {
      const bf16_t* p = &vts[ct * 16 + n][kh * 16];
      fb.q[0] = *(const u32x4*)p;
      fb.q[1] = *(const u32x4*)(p + 8);
      Oa[ct].v = WMMA_BF16(fp.v, fb.v, Oa[ct].v);
    }
    __syncthreads();
  }

  for (int ct = 0; ct < 4; ++ct)
    for (int i = 0; i < 8; ++i) {
      int qrow = q0 + kh * 8 + i;
      int col = h * DK + ct * 16 + n;
      ctx[(size_t)(b * S + qrow) * D + col] = (bf16_t)(Oa[ct].e[i] / l_run[i]);
    }
}

// ------------------------------- launch -------------------------------------
extern "C" void kernel_launch(void* const* d_in, const int* in_sizes, int n_in,
                              void* d_out, int out_size, void* d_ws, size_t ws_size,
                              hipStream_t stream) {
  (void)in_sizes; (void)n_in; (void)out_size; (void)ws_size;
  constexpr int Bn = 16, S = 512, D = 1024, H = 16, DFF = 4096;
  constexpr int M = Bn * S;

  const float* x      = (const float*)d_in[0];
  const float* timein = (const float*)d_in[1];
  const float* Wq = (const float*)d_in[2];  const float* bq = (const float*)d_in[3];
  const float* Wk = (const float*)d_in[4];  const float* bk = (const float*)d_in[5];
  const float* Wt = (const float*)d_in[6];  const float* bt = (const float*)d_in[7];
  const float* Wv = (const float*)d_in[8];  const float* bv = (const float*)d_in[9];
  const float* Wo = (const float*)d_in[10]; const float* bo = (const float*)d_in[11];
  const float* W1 = (const float*)d_in[12]; const float* b1 = (const float*)d_in[13];
  const float* W2 = (const float*)d_in[14]; const float* b2 = (const float*)d_in[15];
  const float* ln1g = (const float*)d_in[16]; const float* ln1b = (const float*)d_in[17];
  const float* ln2g = (const float*)d_in[18]; const float* ln2b = (const float*)d_in[19];
  const float* mlog = (const float*)d_in[20];
  const int*   mask = (const int*)d_in[21];

  char* ws = (char*)d_ws; size_t off = 0;
  auto alloc = [&](size_t bytes) -> void* {
    void* p = ws + off; off += (bytes + 255) & ~size_t(255); return p;
  };

  bf16_t* wq_b = (bf16_t*)alloc(sizeof(bf16_t) * D * D);      // weights stored [N][K]
  bf16_t* wk_b = (bf16_t*)alloc(sizeof(bf16_t) * D * D);
  bf16_t* wt_b = (bf16_t*)alloc(sizeof(bf16_t) * D * D);
  bf16_t* wv_b = (bf16_t*)alloc(sizeof(bf16_t) * D * D);
  bf16_t* wo_b = (bf16_t*)alloc(sizeof(bf16_t) * D * D);
  bf16_t* w1_b = (bf16_t*)alloc(sizeof(bf16_t) * D * DFF);
  bf16_t* w2_b = (bf16_t*)alloc(sizeof(bf16_t) * DFF * D);
  bf16_t* time_b = (bf16_t*)alloc(sizeof(bf16_t) * (size_t)M * D);
  bf16_t* xn_b   = (bf16_t*)alloc(sizeof(bf16_t) * (size_t)M * D);
  bf16_t* q_b    = (bf16_t*)alloc(sizeof(bf16_t) * (size_t)M * D);
  bf16_t* k1_b   = (bf16_t*)alloc(sizeof(bf16_t) * (size_t)M * D);
  bf16_t* k2_b   = (bf16_t*)alloc(sizeof(bf16_t) * (size_t)M * D);
  bf16_t* v_b    = (bf16_t*)alloc(sizeof(bf16_t) * (size_t)M * D);
  bf16_t* ctx_b  = (bf16_t*)alloc(sizeof(bf16_t) * (size_t)M * D);
  float*  x1_f   = (float*)alloc(sizeof(float) * (size_t)M * D);
  bf16_t* xn2_b  = (bf16_t*)alloc(sizeof(bf16_t) * (size_t)M * D);
  bf16_t* h_b    = (bf16_t*)alloc(sizeof(bf16_t) * (size_t)M * DFF);
  float*  sig_f  = (float*)alloc(sizeof(float) * S * S);

  auto cvtT = [&](const float* s, bf16_t* d, int K, int N) {
    dim3 g(K / 32, N / 32);
    cvt_transpose_k<<<g, 256, 0, stream>>>(s, d, K, N);
  };
  cvtT(Wq, wq_b, D, D); cvtT(Wk, wk_b, D, D); cvtT(Wt, wt_b, D, D);
  cvtT(Wv, wv_b, D, D); cvtT(Wo, wo_b, D, D);
  cvtT(W1, w1_b, D, DFF); cvtT(W2, w2_b, DFF, D);
  cvt_f32_bf16_k<<<(M * D + 255) / 256, 256, 0, stream>>>(timein, time_b, M * D);
  sigmoid_k<<<(S * S + 255) / 256, 256, 0, stream>>>(mlog, sig_f, S * S);

  // LN1
  ln_k<<<M, 256, 0, stream>>>(x, ln1g, ln1b, xn_b, D);

  // Projections (bf16 out)
  dim3 gD(D / 128, M / 128);
  gemm_bf16_k<0><<<gD, 256, 0, stream>>>(xn_b,   wq_b, bq, nullptr, nullptr, q_b,  M, D, D);
  gemm_bf16_k<0><<<gD, 256, 0, stream>>>(xn_b,   wk_b, bk, nullptr, nullptr, k1_b, M, D, D);
  gemm_bf16_k<0><<<gD, 256, 0, stream>>>(time_b, wt_b, bt, nullptr, nullptr, k2_b, M, D, D);
  gemm_bf16_k<0><<<gD, 256, 0, stream>>>(xn_b,   wv_b, bv, nullptr, nullptr, v_b,  M, D, D);

  // Fused attention
  dim3 gA(S / 128, H, Bn);
  attn_k<<<gA, 256, 0, stream>>>(q_b, k1_b, k2_b, v_b, mask, sig_f, ctx_b);

  // Output projection + residual (f32)
  gemm_bf16_k<2><<<gD, 256, 0, stream>>>(ctx_b, wo_b, bo, x, x1_f, nullptr, M, D, D);

  // LN2 + FFN
  ln_k<<<M, 256, 0, stream>>>(x1_f, ln2g, ln2b, xn2_b, D);
  dim3 gF(DFF / 128, M / 128);
  gemm_bf16_k<1><<<gF, 256, 0, stream>>>(xn2_b, w1_b, b1, nullptr, nullptr, h_b, M, DFF, D);
  gemm_bf16_k<2><<<gD, 256, 0, stream>>>(h_b, w2_b, b2, x1_f, (float*)d_out, nullptr, M, D, DFF);
}
